// RingAttention_62947040690412
// MI455X (gfx1250) — matrix operand order repaired and checked by reading, not compiled, not run on hardware
//
#include <hip/hip_runtime.h>

// ---------------------------------------------------------------------------
// MHA forward for MI455X (gfx1250, wave32, WMMA).
// Pipeline: f32->bf16 convert | QKV GEMM (wmma bf16, 32x64 wave tile,
// ping-pong prefetch) | flash attention (wmma, K prefetch + V latency hidden
// under softmax) | output GEMM. All accumulation in f32.
// ---------------------------------------------------------------------------

typedef __bf16 bf16;
typedef __attribute__((ext_vector_type(16))) __bf16 bf16x16;
typedef __attribute__((ext_vector_type(8)))  __bf16 bf16x8;
typedef __attribute__((ext_vector_type(4)))  __bf16 bf16x4;
typedef __attribute__((ext_vector_type(8)))  float  v8f;
typedef __attribute__((ext_vector_type(4)))  float  f32x4;

constexpr int Bsz  = 2;
constexpr int Sseq = 4096;
constexpr int Dm   = 1024;
constexpr int Hh   = 16;
constexpr int HD   = 64;          // head dim
constexpr int BH   = Bsz * Hh;    // 32 attention problems

#define WMMA_BF16(a, b, c) \
  __builtin_amdgcn_wmma_f32_16x16x32_bf16(false, (a), false, (b), (short)0, (c), false, false)

__device__ __forceinline__ bf16x16 mk_frag(bf16x8 lo, bf16x8 hi) {
  bf16x16 a;
#pragma unroll
  for (int i = 0; i < 8; ++i) { a[i] = lo[i]; a[i + 8] = hi[i]; }
  return a;
}

// A fragment (16x32, bf16, row-major, row stride ld): lane m=lane&15,
// half=lane>>4; regs 0..3 = k half*8..+7, regs 4..7 = k 16+half*8..+7.
__device__ __forceinline__ bf16x16 load_a_frag(const bf16* base, int ld) {
  const int lane = threadIdx.x & 31;
  const int m = lane & 15, hf = lane >> 4;
  const bf16* p = base + m * ld + hf * 8;
  bf16x8 lo = *(const bf16x8*)(p);
  bf16x8 hi = *(const bf16x8*)(p + 16);
  return mk_frag(lo, hi);
}

// B fragment (32x16) with B[k][n] = Row[n][k] (row-major Row, stride ld):
// lane n=lane&15, half=lane>>4; regs 0..3 = k half*8..+7, regs 4..7 = k 16+half*8..+7.
__device__ __forceinline__ bf16x16 load_b_frag(const bf16* base, int ld) {
  const int lane = threadIdx.x & 31;
  const int n = lane & 15, hf = lane >> 4;
  const bf16* p = base + n * ld + hf * 8;
  bf16x8 lo = *(const bf16x8*)(p);
  bf16x8 hi = *(const bf16x8*)(p + 16);
  return mk_frag(lo, hi);
}

// ---------------------------------------------------------------------------
// Kernel 0: f32 -> bf16 conversion, vectorized (b128 loads / b64 stores)
// ---------------------------------------------------------------------------
__global__ void f32_to_bf16_kernel(const float* __restrict__ in,
                                   bf16* __restrict__ out, int n4) {
  int i = blockIdx.x * blockDim.x + threadIdx.x;
  const int stride = gridDim.x * blockDim.x;
  for (; i < n4; i += stride) {
    f32x4 v = ((const f32x4*)in)[i];
    bf16x4 o;
#pragma unroll
    for (int j = 0; j < 4; ++j) o[j] = (bf16)v[j];
    ((bf16x4*)out)[i] = o;
  }
}

// ---------------------------------------------------------------------------
// Shared GEMM core: one wave computes a 32x64 tile of A(MxK) @ W^T where W is
// (N x K) row-major. Ping-pong double buffering of fragments so prefetch for
// K-step k+32 is in flight while WMMAs for step k execute (no loadcnt-0
// stalls). Prefetch may overrun K by 32 elements -> workspace slack covers.
// ---------------------------------------------------------------------------
__device__ __forceinline__ void gemm_wave_32x64(const bf16* aBase, const bf16* wBase,
                                                int ldA, int ldW, int Kdim,
                                                v8f (&acc)[2][4]) {
  bf16x16 ab[2][2], bb[2][4];
  ab[0][0] = load_a_frag(aBase, ldA);
  ab[0][1] = load_a_frag(aBase + (size_t)16 * ldA, ldA);
#pragma unroll
  for (int nt = 0; nt < 4; ++nt)
    bb[0][nt] = load_b_frag(wBase + (size_t)(nt * 16) * ldW, ldW);

#pragma unroll 1
  for (int kk = 0; kk < Kdim; kk += 64) {
#pragma unroll
    for (int ph = 0; ph < 2; ++ph) {
      const int knext = kk + ph * 32 + 32;
      // prefetch next K-step into the alternate buffer (issues before WMMAs)
      ab[ph ^ 1][0] = load_a_frag(aBase + knext, ldA);
      ab[ph ^ 1][1] = load_a_frag(aBase + (size_t)16 * ldA + knext, ldA);
#pragma unroll
      for (int nt = 0; nt < 4; ++nt)
        bb[ph ^ 1][nt] = load_b_frag(wBase + (size_t)(nt * 16) * ldW + knext, ldW);
      // compute current K-step (fragments loaded one phase ago)
#pragma unroll
      for (int nt = 0; nt < 4; ++nt) {
        acc[0][nt] = WMMA_BF16(ab[ph][0], bb[ph][nt], acc[0][nt]);
        acc[1][nt] = WMMA_BF16(ab[ph][1], bb[ph][nt], acc[1][nt]);
      }
    }
  }
}

// ---------------------------------------------------------------------------
// Kernel 1: QKV projection. C = X(8192x1024) @ Wqkv^T(1024x3072).
// Workgroup = 8 waves x (32x64) = 256x64 tile. Epilogue scatters to Q
// (pre-scaled by 1/sqrt(64)), K (B*H,S,64) and V^T (B*H,64,S) bf16 layouts.
// ---------------------------------------------------------------------------
__global__ __launch_bounds__(256) void qkv_gemm_kernel(
    const bf16* __restrict__ X, const bf16* __restrict__ W,
    bf16* __restrict__ Q, bf16* __restrict__ Kd, bf16* __restrict__ Vt) {
  const int wave = threadIdx.x >> 5;
  const int lane = threadIdx.x & 31;
  const int hf = lane >> 4, col = lane & 15;
  const int gm = blockIdx.y * 256 + wave * 32;   // row in [0, 8192)
  const int gn = blockIdx.x * 64;                // col in [0, 3072)

  v8f acc[2][4] = {};
  gemm_wave_32x64(X + (size_t)gm * Dm, W + (size_t)gn * Dm, Dm, Dm, Dm, acc);

#pragma unroll
  for (int mf = 0; mf < 2; ++mf) {
#pragma unroll
    for (int nt = 0; nt < 4; ++nt) {
#pragma unroll
      for (int r = 0; r < 8; ++r) {
        const int row  = gm + mf * 16 + r + 8 * hf;   // b*S + s
        const int colg = gn + nt * 16 + col;          // [0, 3072)
        const int b = row >> 12, s = row & (Sseq - 1);
        const int which = colg >> 10;                 // 0=Q 1=K 2=V
        const int d = colg & (Dm - 1);
        const int h = d >> 6, di = d & 63;
        const int bh = b * Hh + h;
        const float v = acc[mf][nt][r];
        if (which == 0)
          Q[((size_t)bh * Sseq + s) * HD + di] = (bf16)(v * 0.125f);
        else if (which == 1)
          Kd[((size_t)bh * Sseq + s) * HD + di] = (bf16)v;
        else
          Vt[((size_t)bh * HD + di) * Sseq + s] = (bf16)v;   // transposed
      }
    }
  }
}

__device__ __forceinline__ void load_kfrags(const bf16* kbase, int kb, bf16x16 (&kf)[4]) {
  kf[0] = load_b_frag(kbase + (size_t)kb * HD, HD);
  kf[1] = load_b_frag(kbase + (size_t)kb * HD + 32, HD);
  kf[2] = load_b_frag(kbase + (size_t)(kb + 16) * HD, HD);
  kf[3] = load_b_frag(kbase + (size_t)(kb + 16) * HD + 32, HD);
}

__device__ __forceinline__ void load_vfrags(const bf16* vbase, int kb, bf16x16 (&vf)[4]) {
#pragma unroll
  for (int nt = 0; nt < 4; ++nt)
    vf[nt] = load_b_frag(vbase + (size_t)(nt * 16) * Sseq + kb, Sseq);
}

// ---------------------------------------------------------------------------
// Kernel 2: flash attention. Workgroup = 128 queries of one (b,h); each wave
// owns 16 queries, streams 32 keys/phase. K fragments are ping-pong
// prefetched one phase ahead (consumed immediately by score WMMAs); V
// fragments are issued before the softmax so exp/shuffle VALU hides their
// latency. P transposed to A-layout through a per-wave LDS tile.
// ---------------------------------------------------------------------------
__global__ __launch_bounds__(256) void attn_kernel(
    const bf16* __restrict__ Q, const bf16* __restrict__ Kd,
    const bf16* __restrict__ Vt, bf16* __restrict__ A) {
  __shared__ alignas(16) bf16 pbuf[8][16 * 32];

  const int wave = threadIdx.x >> 5;
  const int lane = threadIdx.x & 31;
  const int hf = lane >> 4, col = lane & 15;
  const int bh = blockIdx.y;
  const int q0 = blockIdx.x * 128 + wave * 16;

  const bf16* qb = Q + ((size_t)bh * Sseq + q0) * HD;
  const bf16x16 aq0 = load_a_frag(qb, HD);        // head dims 0..31
  const bf16x16 aq1 = load_a_frag(qb + 32, HD);   // head dims 32..63

  float m[8], l[8];
  v8f ov[4] = {};
#pragma unroll
  for (int r = 0; r < 8; ++r) { m[r] = -1e30f; l[r] = 0.0f; }

  const bf16* kbase = Kd + (size_t)bh * Sseq * HD;
  const bf16* vbase = Vt + (size_t)bh * HD * Sseq;
  bf16* pw = &pbuf[wave][0];

  bf16x16 kf[2][4];
  load_kfrags(kbase, 0, kf[0]);

#pragma unroll 1
  for (int kb = 0; kb < Sseq; kb += 64) {
#pragma unroll
    for (int ph = 0; ph < 2; ++ph) {
      const int kcur = kb + ph * 32;
      // prefetch next K block (final-phase overrun lands in workspace slack)
      load_kfrags(kbase, kcur + 32, kf[ph ^ 1]);
      // issue current V loads now; softmax below hides their latency
      bf16x16 vf[4];
      load_vfrags(vbase, kcur, vf);

      // ---- scores: 16 queries x 32 keys (Q pre-scaled) ----
      v8f s0 = {}, s1 = {};
      s0 = WMMA_BF16(aq0, kf[ph][0], s0);
      s0 = WMMA_BF16(aq1, kf[ph][1], s0);
      s1 = WMMA_BF16(aq0, kf[ph][2], s1);
      s1 = WMMA_BF16(aq1, kf[ph][3], s1);

      // ---- online softmax (row r+8*hf lives in one 16-lane half) ----
      float alpha[8];
#pragma unroll
      for (int r = 0; r < 8; ++r) {
        float v = fmaxf(s0[r], s1[r]);
        v = fmaxf(v, __shfl_xor(v, 1));
        v = fmaxf(v, __shfl_xor(v, 2));
        v = fmaxf(v, __shfl_xor(v, 4));
        v = fmaxf(v, __shfl_xor(v, 8));
        const float mn = fmaxf(m[r], v);
        alpha[r] = __expf(m[r] - mn);
        m[r] = mn;
        const float p0 = __expf(s0[r] - mn);
        const float p1 = __expf(s1[r] - mn);
        s0[r] = p0; s1[r] = p1;
        float rs = p0 + p1;
        rs += __shfl_xor(rs, 1);
        rs += __shfl_xor(rs, 2);
        rs += __shfl_xor(rs, 4);
        rs += __shfl_xor(rs, 8);
        l[r] = l[r] * alpha[r] + rs;
      }

      // ---- transpose P (C-layout) -> A-fragment layout through LDS ----
      __syncthreads();
#pragma unroll
      for (int r = 0; r < 8; ++r) {
        const int row = r + 8 * hf;
        pw[row * 32 + col]      = (bf16)s0[r];
        pw[row * 32 + 16 + col] = (bf16)s1[r];
      }
      __syncthreads();
      bf16x16 ap = load_a_frag(pw, 32);

      // ---- O = O*alpha + P @ V ----
#pragma unroll
      for (int nt = 0; nt < 4; ++nt) {
#pragma unroll
        for (int r = 0; r < 8; ++r) ov[nt][r] *= alpha[r];
        ov[nt] = WMMA_BF16(ap, vf[nt], ov[nt]);
      }
    }
  }

  // ---- normalize and write attn in (B, S, H*64) row-major bf16 ----
  const int b = bh >> 4, h = bh & 15;
#pragma unroll
  for (int nt = 0; nt < 4; ++nt) {
#pragma unroll
    for (int r = 0; r < 8; ++r) {
      const int s = q0 + r + 8 * hf;
      const int d = h * 64 + nt * 16 + col;
      A[((size_t)b * Sseq + s) * Dm + d] = (bf16)(ov[nt][r] / l[r]);
    }
  }
}

// ---------------------------------------------------------------------------
// Kernel 3: output projection. out = attn(8192x1024) @ Wout^T -> f32.
// ---------------------------------------------------------------------------
__global__ __launch_bounds__(256) void out_gemm_kernel(
    const bf16* __restrict__ A, const bf16* __restrict__ W,
    float* __restrict__ C) {
  const int wave = threadIdx.x >> 5;
  const int lane = threadIdx.x & 31;
  const int hf = lane >> 4, col = lane & 15;
  const int gm = blockIdx.y * 256 + wave * 32;
  const int gn = blockIdx.x * 64;

  v8f acc[2][4] = {};
  gemm_wave_32x64(A + (size_t)gm * Dm, W + (size_t)gn * Dm, Dm, Dm, Dm, acc);

#pragma unroll
  for (int mf = 0; mf < 2; ++mf)
#pragma unroll
    for (int nt = 0; nt < 4; ++nt)
#pragma unroll
      for (int r = 0; r < 8; ++r)
        C[(size_t)(gm + mf * 16 + r + 8 * hf) * Dm + gn + nt * 16 + col] =
            acc[mf][nt][r];
}

// ---------------------------------------------------------------------------
extern "C" void kernel_launch(void* const* d_in, const int* in_sizes, int n_in,
                              void* d_out, int out_size, void* d_ws, size_t ws_size,
                              hipStream_t stream) {
  const float* x     = (const float*)d_in[0];   // (B,S,D)
  const float* w_qkv = (const float*)d_in[1];   // (3D,D)
  const float* w_out = (const float*)d_in[2];   // (D,D)
  float* out = (float*)d_out;

  char* ws = (char*)d_ws;
  size_t off = 0;
  auto alloc = [&](size_t nbytes) -> void* {
    void* p = ws + off;
    off += (nbytes + 255) & ~(size_t)255;
    return p;
  };

  const int nX    = Bsz * Sseq * Dm;   // 8388608
  const int nWqkv = 3 * Dm * Dm;       // 3145728
  const int nWout = Dm * Dm;           // 1048576

  bf16* xb   = (bf16*)alloc((size_t)nX * 2);
  bf16* wqb  = (bf16*)alloc((size_t)nWqkv * 2);
  bf16* wob  = (bf16*)alloc((size_t)nWout * 2);
  bf16* Qb   = (bf16*)alloc((size_t)BH * Sseq * HD * 2);
  bf16* Kb   = (bf16*)alloc((size_t)BH * Sseq * HD * 2);
  bf16* Vtb  = (bf16*)alloc((size_t)BH * HD * Sseq * 2);
  bf16* Attb = (bf16*)alloc((size_t)nX * 2);
  (void)alloc(1 << 16);   // slack: prefetch overruns land here, never consumed
  (void)ws_size; (void)in_sizes; (void)n_in; (void)out_size;

  f32_to_bf16_kernel<<<1024, 256, 0, stream>>>(x, xb, nX / 4);
  f32_to_bf16_kernel<<<1024, 256, 0, stream>>>(w_qkv, wqb, nWqkv / 4);
  f32_to_bf16_kernel<<<512,  256, 0, stream>>>(w_out, wob, nWout / 4);

  // QKV GEMM: M=8192 (y: 32 tiles of 256), N=3072 (x: 48 tiles of 64)
  qkv_gemm_kernel<<<dim3(48, 32), 256, 0, stream>>>(xb, wqb, Qb, Kb, Vtb);

  // Attention: (S/128, B*H)
  attn_kernel<<<dim3(Sseq / 128, BH), 256, 0, stream>>>(Qb, Kb, Vtb, Attb);

  // Output GEMM: M=8192, N=1024
  out_gemm_kernel<<<dim3(Dm / 64, (Bsz * Sseq) / 256), 256, 0, stream>>>(Attb, wob, out);
}
